// SimpleCellRecurrent_78993038508597
// MI455X (gfx1250) — matrix operand order repaired and checked by reading, not compiled, not run on hardware
//
#include <hip/hip_runtime.h>

typedef __attribute__((ext_vector_type(16))) __bf16 v16bf;
typedef __attribute__((ext_vector_type(8)))  float  v8f;
typedef __attribute__((ext_vector_type(4)))  unsigned int u32x4;

#define PAST  512
#define FUT   256
#define TT    (PAST + FUT)
#define HID   512
#define NTHR  1024      // 32 waves
#define NWAVE 32
#define AKB   16        // K blocks of 32

// a_sw: bf16 h in WMMA-A fragment layout: ((kb*2 + p)*32 + lane')*8 + v8
//   K = kb*32 + 8*hhalf + v8 + 16*p ; lane' = m + 16*hhalf
// block index (kb*2+p) == W for wave W's units j in [16W, 16W+16)
#define ASW_ELEMS (AKB * 2 * 32 * 8)   // 8192 bf16 = 16 KB

#define LOG2E 1.4426950408889634f

// branch-free activations on native v_exp2 / v_rcp
__device__ __forceinline__ float sigm(float x) {
  return __builtin_amdgcn_rcpf(1.0f + __builtin_amdgcn_exp2f(-x * LOG2E));
}
__device__ __forceinline__ float ftanh(float x) {
  // tanh(x) = 2*sigmoid(2x) - 1
  return 2.0f * __builtin_amdgcn_rcpf(1.0f + __builtin_amdgcn_exp2f(-x * (2.0f * LOG2E))) - 1.0f;
}

__device__ __forceinline__ v16bf cvt16(float4 a0, float4 a1, float4 a2, float4 a3) {
  v16bf v;
  v[0]  = (__bf16)a0.x; v[1]  = (__bf16)a0.y; v[2]  = (__bf16)a0.z; v[3]  = (__bf16)a0.w;
  v[4]  = (__bf16)a1.x; v[5]  = (__bf16)a1.y; v[6]  = (__bf16)a1.z; v[7]  = (__bf16)a1.w;
  v[8]  = (__bf16)a2.x; v[9]  = (__bf16)a2.y; v[10] = (__bf16)a2.z; v[11] = (__bf16)a2.w;
  v[12] = (__bf16)a3.x; v[13] = (__bf16)a3.y; v[14] = (__bf16)a3.z; v[15] = (__bf16)a3.w;
  return v;
}

// fp32 -> bf16 weights in per-lane WMMA B-fragment layout, gate-grouped:
// frag = nb*16 + kb ; nb = W*4 + gate ; row n = gate*512 + 16*W + (lane&15)
// value[v] = bf16( Whh[n][kb*32 + 16*(lane>>4) + v] )
__global__ void swizzle_weights(const float* __restrict__ Whh_p,
                                const float* __restrict__ Whh_f,
                                __bf16* __restrict__ out) {
  int id   = blockIdx.x * blockDim.x + threadIdx.x;   // 0 .. 131071
  int cell = id >> 16;                                // 0 = past, 1 = future
  int rem  = id & 0xFFFF;
  int frag = rem >> 5;                                // nb*16 + kb
  int lane = rem & 31;
  int nb = frag >> 4, kb = frag & 15;
  int Wv = nb >> 2, g = nb & 3;
  int n     = g * HID + Wv * 16 + (lane & 15);
  int kbase = kb * 32 + 16 * (lane >> 4);
  const float* src = (cell ? Whh_f : Whh_p) + (size_t)n * HID + kbase;
  __bf16* dst = out + (size_t)cell * 2048 * 32 * 16 + ((size_t)frag * 32 + lane) * 16;
#pragma unroll
  for (int v = 0; v < 16; ++v) dst[v] = (__bf16)src[v];
}

template <int USE_SW>
__device__ __forceinline__ void run_phase(
    int tBegin, int tEnd, const int isPast,
    const float* __restrict__ Whh, const __bf16* __restrict__ WswC,
    const float* __restrict__ Wih, const float* __restrict__ bih,
    const float* __restrict__ bhh, const float* __restrict__ Wd, float bdv,
    __bf16* a_sw, const float* in_t, float* x_lds, float* part_o, float* part_x,
    float* __restrict__ out, int b0, float (&cr)[8]) {
  const int tid  = threadIdx.x;
  const int W    = tid >> 5;
  const int lane = tid & 31;
  const int u    = lane & 15;
  const int half = lane >> 4;
  const int j    = 16 * W + u;          // this lane's hidden unit

  // phase-constant per-lane params (registers)
  float wih4[4], bia4[4];
#pragma unroll
  for (int g = 0; g < 4; ++g) {
    wih4[g] = Wih[g * HID + j];
    bia4[g] = bih[g * HID + j] + bhh[g * HID + j];
  }
  const float wdj = Wd[j];

  for (int t = tBegin; t < tEnd; ++t) {
    __syncthreads();   // barrier A: prev-step a_sw/partials visible

    // ---- overlapped with GEMM: reduce prev step's partials (32 threads) ----
    if (t > 0) {
      if (tid < 16) {
        float s = bdv;
#pragma unroll 1
        for (int W2 = 0; W2 < NWAVE; ++W2) s += part_x[W2 * 16 + tid];
        x_lds[tid] = s;                                  // next-x for future steps
      } else if (tid < 32) {
        int m = tid - 16;
        float s = bdv;
#pragma unroll 1
        for (int W2 = 0; W2 < NWAVE; ++W2) s += part_o[W2 * 16 + m];
        out[(size_t)(b0 + m) * TT + (t - 1)] = s;        // decoder output t-1
      }
    }

    // ---------- GEMM: 4 gate-tiles for this wave's 16 units ----------
    v8f acc[4];
#pragma unroll
    for (int g = 0; g < 4; ++g)
#pragma unroll
      for (int r = 0; r < 8; ++r) acc[g][r] = 0.0f;

#pragma unroll 1
    for (int kb = 0; kb < AKB; ++kb) {
      const u32x4* ap = (const u32x4*)a_sw + kb * 64 + half * 32 + u;
      union { u32x4 uu[2]; v16bf v; } au;
      au.uu[0] = ap[0];
      au.uu[1] = ap[16];
      v16bf av = au.v;

#pragma unroll
      for (int g = 0; g < 4; ++g) {
        v16bf bv;
        if (USE_SW) {
          const u32x4* bp =
              (const u32x4*)WswC + (size_t)((W * 4 + g) * 16 + kb) * 64 + lane * 2;
          union { u32x4 uu[2]; v16bf v; } bu;
          bu.uu[0] = __builtin_nontemporal_load(bp);
          bu.uu[1] = __builtin_nontemporal_load(bp + 1);
          bv = bu.v;
        } else {
          int n0 = g * HID + W * 16 + u;
          const float* wr = Whh + (size_t)n0 * HID + kb * 32 + 16 * half;
          float4 b0v = *(const float4*)(wr);
          float4 b1v = *(const float4*)(wr + 4);
          float4 b2v = *(const float4*)(wr + 8);
          float4 b3v = *(const float4*)(wr + 12);
          bv = cvt16(b0v, b1v, b2v, b3v);
        }
        acc[g] = __builtin_amdgcn_wmma_f32_16x16x32_bf16(
            false, av, false, bv, (short)0, acc[g], false, false);
      }
    }

    __syncthreads();   // barrier B: all a_sw reads done; x_lds ready

    // ---------- LSTM update straight from accumulators ----------
    // lane owns unit j for rows m = r + 8*half, r = 0..7
    float xv[8];
    if (isPast) {
      float4 x0 = *(const float4*)(in_t + t * 16 + 8 * half);
      float4 x1 = *(const float4*)(in_t + t * 16 + 8 * half + 4);
      xv[0] = x0.x; xv[1] = x0.y; xv[2] = x0.z; xv[3] = x0.w;
      xv[4] = x1.x; xv[5] = x1.y; xv[6] = x1.z; xv[7] = x1.w;
    } else {
      float4 x0 = *(const float4*)(x_lds + 8 * half);
      float4 x1 = *(const float4*)(x_lds + 8 * half + 4);
      xv[0] = x0.x; xv[1] = x0.y; xv[2] = x0.z; xv[3] = x0.w;
      xv[4] = x1.x; xv[5] = x1.y; xv[6] = x1.z; xv[7] = x1.w;
    }

    float po[8], px[8];
#pragma unroll
    for (int r = 0; r < 8; ++r) {
      float xr = xv[r];
      float pi = acc[0][r] + xr * wih4[0] + bia4[0];
      float pf = acc[1][r] + xr * wih4[1] + bia4[1];
      float pg = acc[2][r] + xr * wih4[2] + bia4[2];
      float pq = acc[3][r] + xr * wih4[3] + bia4[3];
      float c  = sigm(pf) * cr[r] + sigm(pi) * ftanh(pg);
      float h  = sigm(pq) * ftanh(c);
      cr[r] = c;
      int lanep = r + 8 * half + 16 * ((u >> 3) & 1);
      a_sw[(size_t)(W * 32 + lanep) * 8 + (u & 7)] = (__bf16)h;
      po[r] = h * wdj;
      px[r] = c * wdj;
    }
    // butterfly over the 16 lanes sharing the same m-set
#pragma unroll
    for (int mask = 1; mask < 16; mask <<= 1)
#pragma unroll
      for (int r = 0; r < 8; ++r) {
        po[r] += __shfl_xor(po[r], mask, 32);
        px[r] += __shfl_xor(px[r], mask, 32);
      }
    if (u == 0) {        // lanes 0 (m=0..7) and 16 (m=8..15)
#pragma unroll
      for (int r = 0; r < 8; ++r) {
        part_o[W * 16 + 8 * half + r] = po[r];
        part_x[W * 16 + 8 * half + r] = px[r];
      }
    }
  }
}

template <int USE_SW>
__global__ __launch_bounds__(NTHR)
void lstm_main(const float* __restrict__ in_seq,
               const float* __restrict__ Wih_p, const float* __restrict__ Whh_p,
               const float* __restrict__ bih_p, const float* __restrict__ bhh_p,
               const float* __restrict__ Wih_f, const float* __restrict__ Whh_f,
               const float* __restrict__ bih_f, const float* __restrict__ bhh_f,
               const float* __restrict__ Wd,    const float* __restrict__ bd,
               const __bf16* __restrict__ Wsw,
               float* __restrict__ out) {
  extern __shared__ char smem[];
  __bf16* a_sw  = (__bf16*)smem;                    // 16 KB bf16 h (A-swizzled)
  float* in_t   = (float*)(smem + ASW_ELEMS * 2);   // [PAST][16] transposed input
  float* x_lds  = in_t + 16 * PAST;                 // [16]
  float* part_o = x_lds + 16;                       // [32][16]
  float* part_x = part_o + NWAVE * 16;              // [32][16]

  const int tid = threadIdx.x;
  const int b0  = blockIdx.x * 16;   // batch tile base

  // ---- stage input (transposed) & init state ----
  for (int i = tid; i < 16 * PAST; i += NTHR)
    in_t[i] = in_seq[(size_t)(b0 + (i & 15)) * PAST + (i >> 4)];
  {
    const __bf16 h0 = (__bf16)0.01f;
    for (int i = tid; i < ASW_ELEMS; i += NTHR) a_sw[i] = h0;
  }
  if (tid < 16) x_lds[tid] = 0.0f;

  float cr[8];
#pragma unroll
  for (int r = 0; r < 8; ++r) cr[r] = 0.01f;

  const float bdv = bd[0];
  const __bf16* Wsw_p = Wsw;
  const __bf16* Wsw_f = Wsw + (size_t)2048 * 32 * 16;

  run_phase<USE_SW>(0, PAST, 1, Whh_p, Wsw_p, Wih_p, bih_p, bhh_p, Wd, bdv,
                    a_sw, in_t, x_lds, part_o, part_x, out, b0, cr);
  run_phase<USE_SW>(PAST, TT, 0, Whh_f, Wsw_f, Wih_f, bih_f, bhh_f, Wd, bdv,
                    a_sw, in_t, x_lds, part_o, part_x, out, b0, cr);

  // epilogue: final step's decoder output
  __syncthreads();
  if (tid >= 16 && tid < 32) {
    int m = tid - 16;
    float s = bdv;
#pragma unroll 1
    for (int W2 = 0; W2 < NWAVE; ++W2) s += part_o[W2 * 16 + m];
    out[(size_t)(b0 + m) * TT + (TT - 1)] = s;
  }
}

extern "C" void kernel_launch(void* const* d_in, const int* in_sizes, int n_in,
                              void* d_out, int out_size, void* d_ws, size_t ws_size,
                              hipStream_t stream) {
  const float* in_seq = (const float*)d_in[0];
  // d_in[1] = future_n (compile-time constant 256, unused)
  const float* Wih_p = (const float*)d_in[2];
  const float* Whh_p = (const float*)d_in[3];
  const float* bih_p = (const float*)d_in[4];
  const float* bhh_p = (const float*)d_in[5];
  const float* Wih_f = (const float*)d_in[6];
  const float* Whh_f = (const float*)d_in[7];
  const float* bih_f = (const float*)d_in[8];
  const float* bhh_f = (const float*)d_in[9];
  const float* Wd    = (const float*)d_in[10];
  const float* bd    = (const float*)d_in[11];
  float* out = (float*)d_out;

  __bf16* Wsw = (__bf16*)d_ws;
  const size_t need = (size_t)2 * 2048 * 32 * 16 * sizeof(__bf16);  // 4 MB
  const int use_sw = (ws_size >= need) ? 1 : 0;

  size_t lds = (size_t)ASW_ELEMS * 2 +
               (size_t)(16 * PAST + 16 + NWAVE * 16 * 2) * sizeof(float); // ~53 KB

  if (use_sw) {
    swizzle_weights<<<512, 256, 0, stream>>>(Whh_p, Whh_f, Wsw);
    (void)hipFuncSetAttribute(reinterpret_cast<const void*>(lstm_main<1>),
                              hipFuncAttributeMaxDynamicSharedMemorySize, (int)lds);
    lstm_main<1><<<16, NTHR, lds, stream>>>(in_seq, Wih_p, Whh_p, bih_p, bhh_p,
                                            Wih_f, Whh_f, bih_f, bhh_f, Wd, bd,
                                            Wsw, out);
  } else {
    (void)hipFuncSetAttribute(reinterpret_cast<const void*>(lstm_main<0>),
                              hipFuncAttributeMaxDynamicSharedMemorySize, (int)lds);
    lstm_main<0><<<16, NTHR, lds, stream>>>(in_seq, Wih_p, Whh_p, bih_p, bhh_p,
                                            Wih_f, Whh_f, bih_f, bhh_f, Wd, bd,
                                            Wsw, out);
  }
}
